// TopKWindowAttentionLayer_39917426049202
// MI455X (gfx1250) — matrix operand order repaired and checked by reading, not compiled, not run on hardware
//
#include <hip/hip_runtime.h>

// ---------------------------------------------------------------------------
// Types & WMMA helpers (gfx1250, wave32, v_wmma_f32_16x16x32_bf16)
// ---------------------------------------------------------------------------
typedef __bf16 bhalf;
typedef bhalf v16bf __attribute__((ext_vector_type(16)));
typedef float v8f   __attribute__((ext_vector_type(8)));
typedef unsigned int u32x4 __attribute__((ext_vector_type(4)));
typedef int i32x4 __attribute__((ext_vector_type(4)));
typedef int i32x8 __attribute__((ext_vector_type(8)));

union BF16x16 { v16bf v; unsigned int u[8]; };

__device__ __forceinline__ unsigned int pack_bf(bhalf lo, bhalf hi) {
    union { bhalf h[2]; unsigned int u; } x;
    x.h[0] = lo; x.h[1] = hi;
    return x.u;
}

__device__ __forceinline__ v8f wmma_bf16(const BF16x16& a, const BF16x16& b, v8f c) {
    return __builtin_amdgcn_wmma_f32_16x16x32_bf16(
        /*neg_a=*/false, a.v, /*neg_b=*/false, b.v,
        /*c_mod=*/(short)0, c, /*reuse_a=*/false, /*reuse_b=*/false);
}

// A-fragment (16x32 bf16): lane L holds row M=L&15; dword d holds K pair
//   k = 2*(d&3) + 8*(L>>4) + 16*(d>>2)  (dwords 0-3 and 4-7 are each 16B runs)
__device__ __forceinline__ int a_kofs(int d, int half) {
    return 2 * (d & 3) + 8 * half + 16 * (d >> 2);
}
// B-fragment (32x16 bf16): lane L holds col N=L&15; dword d holds K pair
//   k = 2*(d&3) + 8*(L>>4) + 16*(d>>2)
__device__ __forceinline__ int b_kofs(int d, int half) {
    return 2 * (d & 3) + 8 * half + 16 * (d >> 2);
}

// ---------------------------------------------------------------------------
// Tensor Data Mover: 2-D tile load Global -> LDS, hardware row padding.
// D# group0: count=1 | lds_addr | global_addr | type=2
// D# group1: data_size=2B, pad_enable, pad_interval/amount, dims/strides
// Issued by one wave; completion via TENSORcnt.
// ---------------------------------------------------------------------------
__device__ __forceinline__ void tdm_load_2d(unsigned lds_off, const void* gptr,
                                            unsigned tile_w, unsigned tile_h,
                                            unsigned stride_elems,
                                            unsigned pad_interval_code,
                                            unsigned pad_amount_code) {
    unsigned long long ga = (unsigned long long)(size_t)gptr;
    u32x4 g0;
    g0.x = 1u;                                     // count = 1 valid descriptor
    g0.y = lds_off;                                // LDS byte address
    g0.z = (unsigned)ga;                           // global_addr[31:0]
    g0.w = (unsigned)(ga >> 32) | (2u << 30);      // global_addr[56:32] | type=2
    i32x8 g1;
    g1[0] = (int)((1u << 16)                       // data_size = 2 bytes
                | (1u << 20)                       // pad_enable
                | (pad_interval_code << 22)
                | (pad_amount_code << 25));
    g1[1] = (int)((tile_w & 0xFFFFu) << 16);       // tensor_dim0[15:0] (= tile_w)
    g1[2] = (int)(((tile_w >> 16) & 0xFFFFu)       // tensor_dim0[31:16]
                | ((tile_h & 0xFFFFu) << 16));     // tensor_dim1[15:0] (= tile_h)
    g1[3] = (int)(((tile_h >> 16) & 0xFFFFu)       // tensor_dim1[31:16]
                | (tile_w << 16));                 // tile_dim0
    g1[4] = (int)(tile_h & 0xFFFFu);               // tile_dim1 | tile_dim2=0
    g1[5] = (int)stride_elems;                     // tensor_dim0_stride[31:0]
    g1[6] = 0;                                     // stride hi | dim1_stride lo
    g1[7] = 0;
    i32x4 z4 = {0, 0, 0, 0};
#if defined(__clang_major__) && __clang_major__ >= 23
    i32x8 z8 = {0, 0, 0, 0, 0, 0, 0, 0};
    __builtin_amdgcn_tensor_load_to_lds(g0, g1, z4, z4, z8, 0);
#else
    __builtin_amdgcn_tensor_load_to_lds(g0, g1, z4, z4, 0);
#endif
}

// Problem constants
#define BS 2
#define D 256
#define HW 12544            // 112*112
#define PIX (BS * HW)       // 25088
#define NW 256              // 16*16 windows per batch
#define WW 49               // 7*7
#define NH 8
#define DH 32
#define SKV 648             // 8*49 + 256
#define SKV_PAD 672         // 21 chunks of 32

// ---------------------------------------------------------------------------
// Row-permutation maps: pixel-major <-> window-major
// ---------------------------------------------------------------------------
__global__ void k_init_maps(int* p2w, int* w2p) {
    int p = blockIdx.x * 256 + threadIdx.x;
    if (p >= PIX) return;
    int b = p / HW, hw = p % HW, h = hw / 112, w = hw % 112;
    int wr = b * HW + ((h / 7) * 16 + (w / 7)) * WW + (h % 7) * 7 + (w % 7);
    p2w[p] = wr;
    w2p[wr] = p;
}

// ---------------------------------------------------------------------------
// NCHW -> [pixel, channel], fp32 and/or bf16 (bf16 can land in a wider buffer)
// ---------------------------------------------------------------------------
__global__ void k_transpose_in(const float* __restrict__ in, float* outf,
                               bhalf* outb, int ldo, int coff) {
    __shared__ float tile[32][33];
    int b = blockIdx.z, c0 = blockIdx.y * 32, p0 = blockIdx.x * 32;
    int tx = threadIdx.x;
    for (int ty = threadIdx.y; ty < 32; ty += 8)
        tile[ty][tx] = in[((size_t)(b * D + c0 + ty)) * HW + p0 + tx];
    __syncthreads();
    for (int ty = threadIdx.y; ty < 32; ty += 8) {
        float v = tile[tx][ty];
        size_t row = (size_t)b * HW + p0 + ty;
        if (outf) outf[row * D + c0 + tx] = v;
        if (outb) outb[row * (size_t)ldo + coff + c0 + tx] = (bhalf)v;
    }
}

// [pixel, channel] fp32 -> NCHW fp32
__global__ void k_transpose_out(const float* __restrict__ in, float* __restrict__ out) {
    __shared__ float tile[32][33];
    int b = blockIdx.z, c0 = blockIdx.y * 32, p0 = blockIdx.x * 32;
    int tx = threadIdx.x;
    for (int ty = threadIdx.y; ty < 32; ty += 8)
        tile[ty][tx] = in[((size_t)b * HW + p0 + ty) * D + c0 + tx];
    __syncthreads();
    for (int ty = threadIdx.y; ty < 32; ty += 8)
        out[((size_t)(b * D + c0 + ty)) * HW + p0 + tx] = tile[tx][ty];
}

__global__ void k_cvt_bf16(const float* __restrict__ in, bhalf* __restrict__ out, int n) {
    int i = blockIdx.x * 256 + threadIdx.x;
    if (i < n) out[i] = (bhalf)in[i];
}

// ---------------------------------------------------------------------------
// Generic bf16 WMMA GEMM with TDM-fed, double-buffered LDS tiles.
// Block = 128 threads (4 waves), tile 64x64, K-step 32; wave 0 issues
// tensor_load_to_lds for the next K-step while all waves compute on the
// current one (TENSORcnt pipelining: wait <=2 keeps the newest pair in
// flight). TDM hardware padding reproduces the bank-conflict-free layouts
// sA[64][40] (64B rows + 16B pad) and sB[32][72] (128B rows + 16B pad).
// flags: bit0 -> bf16 output, bit1 -> relu
// ---------------------------------------------------------------------------
__global__ void __launch_bounds__(128)
k_gemm(const bhalf* __restrict__ A, int lda, const bhalf* __restrict__ B, int ldb,
       void* Cp, int ldc, int K, const int* __restrict__ row_map, int flags) {
    __shared__ bhalf sA[2][64][40];
    __shared__ bhalf sB[2][32][72];
    int tm = blockIdx.x * 64, tn = blockIdx.y * 64;
    int t = threadIdx.x, wave = t >> 5, lane = t & 31;
    int lm = lane & 15, half = lane >> 4;
    int wm = (wave >> 1) * 32, wn = (wave & 1) * 32;
    bool issuer = (t < 32);  // wave 0 drives the TDM

    const bhalf* Abase = A + (size_t)tm * lda;
    const bhalf* Bbase = B + tn;
    int nk = K >> 5;

    if (issuer) {  // prologue: fetch K-step 0 into buffer 0
        tdm_load_2d((unsigned)(size_t)&sA[0][0][0], Abase, 32, 64, (unsigned)lda, 3, 3);
        tdm_load_2d((unsigned)(size_t)&sB[0][0][0], Bbase, 64, 32, (unsigned)ldb, 4, 3);
    }

    v8f acc[2][2] = {};
    for (int ki = 0; ki < nk; ++ki) {
        int buf = ki & 1;
        if (issuer) {
            if (ki + 1 < nk) {  // prefetch next K-step into the other buffer
                tdm_load_2d((unsigned)(size_t)&sA[buf ^ 1][0][0],
                            Abase + (ki + 1) * 32, 32, 64, (unsigned)lda, 3, 3);
                tdm_load_2d((unsigned)(size_t)&sB[buf ^ 1][0][0],
                            Bbase + (size_t)(ki + 1) * 32 * ldb, 64, 32, (unsigned)ldb, 4, 3);
                __builtin_amdgcn_s_wait_tensorcnt(2);  // retire current pair
            } else {
                __builtin_amdgcn_s_wait_tensorcnt(0);
            }
        }
        __syncthreads();  // current buffer ready for all waves

        BF16x16 af[2], bf[2];
        #pragma unroll
        for (int i = 0; i < 2; ++i)
            #pragma unroll
            for (int d = 0; d < 8; ++d)
                af[i].u[d] = *(const unsigned int*)&sA[buf][wm + 16 * i + lm][a_kofs(d, half)];
        #pragma unroll
        for (int j = 0; j < 2; ++j)
            #pragma unroll
            for (int d = 0; d < 8; ++d) {
                int kk = b_kofs(d, half);
                bf[j].u[d] = pack_bf(sB[buf][kk][wn + 16 * j + lm],
                                     sB[buf][kk + 1][wn + 16 * j + lm]);
            }
        #pragma unroll
        for (int i = 0; i < 2; ++i)
            #pragma unroll
            for (int j = 0; j < 2; ++j)
                acc[i][j] = wmma_bf16(af[i], bf[j], acc[i][j]);
        __syncthreads();  // all reads done before buffer is refilled
    }

    int relu = flags & 2, obf = flags & 1;
    #pragma unroll
    for (int i = 0; i < 2; ++i)
        #pragma unroll
        for (int j = 0; j < 2; ++j)
            #pragma unroll
            for (int r = 0; r < 8; ++r) {
                int mrow = tm + wm + 16 * i + r + 8 * half;
                int n = tn + wn + 16 * j + lm;
                int orow = row_map ? row_map[mrow] : mrow;
                float v = acc[i][j][r];
                if (relu) v = fmaxf(v, 0.0f);
                if (obf) ((bhalf*)Cp)[(size_t)orow * ldc + n] = (bhalf)v;
                else     ((float*)Cp)[(size_t)orow * ldc + n] = v;
            }
}

// ---------------------------------------------------------------------------
// Window means over 49 rows (window-major bf16 input)
// ---------------------------------------------------------------------------
__global__ void k_means(const bhalf* __restrict__ in, float* outf, bhalf* outb) {
    int bw = blockIdx.x, c = threadIdx.x;
    const bhalf* base = in + (size_t)bw * WW * D + c;
    float s = 0.0f;
    for (int l = 0; l < WW; ++l) s += (float)base[(size_t)l * D];
    s *= (1.0f / 49.0f);
    if (outf) outf[(size_t)bw * D + c] = s;
    if (outb) outb[(size_t)bw * D + c] = (bhalf)s;
}

// ---------------------------------------------------------------------------
// Coarse sim + iterative top-8 (order irrelevant: softmax sum is
// permutation-invariant over keys)
// ---------------------------------------------------------------------------
__global__ void k_topk(const float* __restrict__ qm, const float* __restrict__ km,
                       int* __restrict__ top) {
    __shared__ float sv[256];
    __shared__ float rv[256];
    __shared__ int   ri[256];
    __shared__ int   sel[8];
    int j = threadIdx.x, bm = blockIdx.x, b = bm >> 8;
    const float* qr = qm + (size_t)bm * D;
    const float* kr = km + ((size_t)(b << 8) + j) * D;
    float s = 0.0f;
    for (int c = 0; c < D; ++c) s += qr[c] * kr[c];
    sv[j] = s;
    __syncthreads();
    for (int it = 0; it < 8; ++it) {
        rv[j] = sv[j]; ri[j] = j;
        __syncthreads();
        for (int st = 128; st > 0; st >>= 1) {
            if (j < st) {
                if (rv[j + st] > rv[j] || (rv[j + st] == rv[j] && ri[j + st] < ri[j])) {
                    rv[j] = rv[j + st]; ri[j] = ri[j + st];
                }
            }
            __syncthreads();
        }
        if (j == 0) { sel[it] = ri[0]; sv[ri[0]] = -3.0e38f; }
        __syncthreads();
    }
    if (j < 8) top[(size_t)bm * 8 + j] = sel[j];
}

// ---------------------------------------------------------------------------
// Flash-style windowed attention. Block = (b,win,head), 128 threads = 4 waves,
// wave w owns query rows [16w, 16w+16). s processed in chunks of 32 (dh=32
// == one bf16 WMMA K step). 4 WMMAs per wave per chunk. K/V gather is an
// indirect two-source gather -> per-lane b128 loads (poor TDM fit).
// ---------------------------------------------------------------------------
__global__ void __launch_bounds__(128)
k_attn(const bhalf* __restrict__ qw, const bhalf* __restrict__ kw,
       const bhalf* __restrict__ vw, const bhalf* __restrict__ kmb,
       const bhalf* __restrict__ vmb, const int* __restrict__ topidx,
       bhalf* __restrict__ msg) {
    __shared__ bhalf sK[32][40];
    __shared__ bhalf sV[32][40];
    __shared__ bhalf sP[4][16][40];
    __shared__ int sTop[8];

    int idx = blockIdx.x;
    int h = idx & 7, win = (idx >> 3) & 255, b = idx >> 11;
    int t = threadIdx.x, wave = t >> 5, lane = t & 31;
    int lm = lane & 15, half = lane >> 4;

    if (t < 8) sTop[t] = topidx[((size_t)(b << 8) + win) * 8 + t];
    __syncthreads();

    // Q fragment (rows >= 49 zeroed -> harmless uniform softmax, never stored)
    BF16x16 qa;
    {
        int row = wave * 16 + lm;
        const bhalf* qb = qw + (((size_t)(b << 8) + win) * WW + row) * D + h * DH;
        #pragma unroll
        for (int d = 0; d < 8; ++d)
            qa.u[d] = (row < WW) ? *(const unsigned int*)(qb + a_kofs(d, half)) : 0u;
    }

    float mrow[8], lrow[8];
    #pragma unroll
    for (int r = 0; r < 8; ++r) { mrow[r] = -1.0e30f; lrow[r] = 0.0f; }
    v8f acc[2] = {};
    const float scale = 0.17677669529663687f;  // 1/sqrt(32)

    for (int c0 = 0; c0 < SKV_PAD; c0 += 32) {
        {   // gather 32 x 32 bf16 K and V rows into LDS
            int r = t >> 2, cg = (t & 3) * 8;
            int sg = c0 + r;
            const bhalf *srcK, *srcV;
            bool valid = sg < SKV;
            if (sg < 392) {
                int j = sg / WW, l = sg - j * WW;
                size_t off = (((size_t)(b << 8) + sTop[j]) * WW + l) * D + h * DH + cg;
                srcK = kw + off; srcV = vw + off;
            } else {
                int s2 = (sg - 392) & 255;  // clamp keeps pointer in-bounds when invalid
                size_t off = ((size_t)(b << 8) + s2) * D + h * DH + cg;
                srcK = kmb + off; srcV = vmb + off;
            }
            int4 z = make_int4(0, 0, 0, 0);
            int4 kk = valid ? *(const int4*)srcK : z;
            int4 vv = valid ? *(const int4*)srcV : z;
            *(int4*)&sK[r][cg] = kk;
            *(int4*)&sV[r][cg] = vv;
        }
        __syncthreads();

        // scores = Q * K^T : B[k=dh][n=s] = sK[n][k] (per-lane contiguous)
        BF16x16 bk[2];
        #pragma unroll
        for (int j = 0; j < 2; ++j)
            #pragma unroll
            for (int d = 0; d < 8; ++d)
                bk[j].u[d] = *(const unsigned int*)&sK[16 * j + lm][b_kofs(d, half)];
        v8f z8 = {};
        v8f sc0 = wmma_bf16(qa, bk[0], z8);
        v8f sc1 = wmma_bf16(qa, bk[1], z8);

        // online softmax (row = r + 8*half, uniform across the 16-lane group)
        bool mask0 = (c0 + lm) >= SKV, mask1 = (c0 + 16 + lm) >= SKV;
        #pragma unroll
        for (int r = 0; r < 8; ++r) {
            float v0 = mask0 ? -1.0e30f : sc0[r] * scale;
            float v1 = mask1 ? -1.0e30f : sc1[r] * scale;
            float mx = fmaxf(v0, v1);
            mx = fmaxf(mx, __shfl_xor(mx, 1, 32));
            mx = fmaxf(mx, __shfl_xor(mx, 2, 32));
            mx = fmaxf(mx, __shfl_xor(mx, 4, 32));
            mx = fmaxf(mx, __shfl_xor(mx, 8, 32));
            float mnew = fmaxf(mrow[r], mx);
            float corr = __expf(mrow[r] - mnew);
            float p0 = __expf(v0 - mnew);
            float p1 = __expf(v1 - mnew);
            float ps = p0 + p1;
            ps += __shfl_xor(ps, 1, 32);
            ps += __shfl_xor(ps, 2, 32);
            ps += __shfl_xor(ps, 4, 32);
            ps += __shfl_xor(ps, 8, 32);
            lrow[r] = lrow[r] * corr + ps;
            mrow[r] = mnew;
            sP[wave][r + 8 * half][lm]      = (bhalf)p0;
            sP[wave][r + 8 * half][16 + lm] = (bhalf)p1;
            acc[0][r] *= corr;
            acc[1][r] *= corr;
        }
        __syncthreads();

        // acc += P (16x32) * V (32x32): B[k=s][n=dh] = sV[k][n]
        BF16x16 pa, bv[2];
        #pragma unroll
        for (int d = 0; d < 8; ++d)
            pa.u[d] = *(const unsigned int*)&sP[wave][lm][a_kofs(d, half)];
        #pragma unroll
        for (int j = 0; j < 2; ++j)
            #pragma unroll
            for (int d = 0; d < 8; ++d) {
                int kk = b_kofs(d, half);
                bv[j].u[d] = pack_bf(sV[kk][16 * j + lm], sV[kk + 1][16 * j + lm]);
            }
        acc[0] = wmma_bf16(pa, bv[0], acc[0]);
        acc[1] = wmma_bf16(pa, bv[1], acc[1]);
        __syncthreads();
    }

    #pragma unroll
    for (int r = 0; r < 8; ++r) {
        int row = wave * 16 + r + 8 * half;
        if (row < WW) {
            float inv = 1.0f / lrow[r];
            size_t base = (((size_t)(b << 8) + win) * WW + row) * D + h * DH;
            msg[base + lm]      = (bhalf)(acc[0][r] * inv);
            msg[base + 16 + lm] = (bhalf)(acc[1][r] * inv);
        }
    }
}

// ---------------------------------------------------------------------------
// LayerNorm(d=256) -> bf16 into right half of concat buffer (ld 512)
// ---------------------------------------------------------------------------
__global__ void k_ln_concat(const float* __restrict__ in, const float* __restrict__ g,
                            const float* __restrict__ be, bhalf* __restrict__ cat) {
    __shared__ float red[256];
    int row = blockIdx.x, c = threadIdx.x;
    float x = in[(size_t)row * D + c];
    red[c] = x; __syncthreads();
    for (int s = 128; s > 0; s >>= 1) { if (c < s) red[c] += red[c + s]; __syncthreads(); }
    float mu = red[0] * (1.0f / 256.0f);
    __syncthreads();
    float dx = x - mu;
    red[c] = dx * dx; __syncthreads();
    for (int s = 128; s > 0; s >>= 1) { if (c < s) red[c] += red[c + s]; __syncthreads(); }
    float var = red[0] * (1.0f / 256.0f);
    float y = dx * rsqrtf(var + 1e-5f) * g[c] + be[c];
    cat[(size_t)row * 512 + 256 + c] = (bhalf)y;
}

// LayerNorm + residual -> fp32
__global__ void k_ln_res(const float* __restrict__ in, const float* __restrict__ xf,
                         const float* __restrict__ g, const float* __restrict__ be,
                         float* __restrict__ out) {
    __shared__ float red[256];
    int row = blockIdx.x, c = threadIdx.x;
    float x = in[(size_t)row * D + c];
    red[c] = x; __syncthreads();
    for (int s = 128; s > 0; s >>= 1) { if (c < s) red[c] += red[c + s]; __syncthreads(); }
    float mu = red[0] * (1.0f / 256.0f);
    __syncthreads();
    float dx = x - mu;
    red[c] = dx * dx; __syncthreads();
    for (int s = 128; s > 0; s >>= 1) { if (c < s) red[c] += red[c + s]; __syncthreads(); }
    float var = red[0] * (1.0f / 256.0f);
    float y = dx * rsqrtf(var + 1e-5f) * g[c] + be[c];
    out[(size_t)row * D + c] = xf[(size_t)row * D + c] + y;
}

// ---------------------------------------------------------------------------
// Host orchestration
// ---------------------------------------------------------------------------
static inline char* bump(char*& p, size_t bytes) {
    char* r = p;
    p += (bytes + 255) & ~(size_t)255;
    return r;
}

extern "C" void kernel_launch(void* const* d_in, const int* in_sizes, int n_in,
                              void* d_out, int out_size, void* d_ws, size_t ws_size,
                              hipStream_t stream) {
    const float* x   = (const float*)d_in[0];
    const float* src = (const float*)d_in[1];
    const float* Wq  = (const float*)d_in[2];
    const float* Wk  = (const float*)d_in[3];
    const float* Wv  = (const float*)d_in[4];
    const float* Wm  = (const float*)d_in[5];
    const float* W1  = (const float*)d_in[6];
    const float* W2  = (const float*)d_in[7];
    const float* g1  = (const float*)d_in[8];
    const float* b1  = (const float*)d_in[9];
    const float* g2  = (const float*)d_in[10];
    const float* b2  = (const float*)d_in[11];
    float* out = (float*)d_out;

    char* ws = (char*)d_ws;
    int*   p2w     = (int*)  bump(ws, (size_t)PIX * 4);
    int*   w2p     = (int*)  bump(ws, (size_t)PIX * 4);
    float* xf      = (float*)bump(ws, (size_t)PIX * D * 4);
    bhalf* cat     = (bhalf*)bump(ws, (size_t)PIX * 512 * 2);
    bhalf* s_bf    = (bhalf*)bump(ws, (size_t)PIX * D * 2);   // reused as msg_bf
    bhalf* Wq_bf   = (bhalf*)bump(ws, (size_t)D * D * 2);
    bhalf* Wk_bf   = (bhalf*)bump(ws, (size_t)D * D * 2);
    bhalf* Wv_bf   = (bhalf*)bump(ws, (size_t)D * D * 2);
    bhalf* Wm_bf   = (bhalf*)bump(ws, (size_t)D * D * 2);
    bhalf* W1_bf   = (bhalf*)bump(ws, (size_t)512 * 512 * 2);
    bhalf* W2_bf   = (bhalf*)bump(ws, (size_t)512 * D * 2);
    bhalf* q_bf    = (bhalf*)bump(ws, (size_t)PIX * D * 2);
    bhalf* k_bf    = (bhalf*)bump(ws, (size_t)PIX * D * 2);
    bhalf* v_bf    = (bhalf*)bump(ws, (size_t)PIX * D * 2);
    float* qmean   = (float*)bump(ws, (size_t)BS * NW * D * 4);
    float* kmean   = (float*)bump(ws, (size_t)BS * NW * D * 4);
    bhalf* kmean_b = (bhalf*)bump(ws, (size_t)BS * NW * D * 2);
    bhalf* vmean_b = (bhalf*)bump(ws, (size_t)BS * NW * D * 2);
    int*   top_idx = (int*)  bump(ws, (size_t)BS * NW * 8 * 4);
    float* msgWm   = (float*)bump(ws, (size_t)PIX * D * 4);   // reused as final pre-transpose
    bhalf* hmid    = (bhalf*)bump(ws, (size_t)PIX * 512 * 2);
    float* msg2    = (float*)bump(ws, (size_t)PIX * D * 4);
    bhalf* msg_bf  = s_bf;  // source bf16 fully consumed before attention writes

    dim3 tb(32, 8);
    dim3 tg(392, 8, BS);  // 392 pixel tiles, 8 channel tiles, 2 batches

    // 1. row maps
    k_init_maps<<<PIX / 256, 256, 0, stream>>>(p2w, w2p);
    // 2. transposes + bf16 conversion (x -> concat left half + fp32 residual)
    k_transpose_in<<<tg, tb, 0, stream>>>(x, xf, cat, 512, 0);
    k_transpose_in<<<tg, tb, 0, stream>>>(src, nullptr, s_bf, D, 0);
    // 3. weight conversion
    k_cvt_bf16<<<(D * D + 255) / 256, 256, 0, stream>>>(Wq, Wq_bf, D * D);
    k_cvt_bf16<<<(D * D + 255) / 256, 256, 0, stream>>>(Wk, Wk_bf, D * D);
    k_cvt_bf16<<<(D * D + 255) / 256, 256, 0, stream>>>(Wv, Wv_bf, D * D);
    k_cvt_bf16<<<(D * D + 255) / 256, 256, 0, stream>>>(Wm, Wm_bf, D * D);
    k_cvt_bf16<<<(512 * 512 + 255) / 256, 256, 0, stream>>>(W1, W1_bf, 512 * 512);
    k_cvt_bf16<<<(512 * D + 255) / 256, 256, 0, stream>>>(W2, W2_bf, 512 * D);
    // 4. projections (output window-major bf16)
    k_gemm<<<dim3(392, 4), 128, 0, stream>>>(cat,  512, Wq_bf, D, q_bf, D, D, p2w, 1);
    k_gemm<<<dim3(392, 4), 128, 0, stream>>>(s_bf, D,   Wk_bf, D, k_bf, D, D, p2w, 1);
    k_gemm<<<dim3(392, 4), 128, 0, stream>>>(s_bf, D,   Wv_bf, D, v_bf, D, D, p2w, 1);
    // 5. window means
    k_means<<<BS * NW, 256, 0, stream>>>(q_bf, qmean, nullptr);
    k_means<<<BS * NW, 256, 0, stream>>>(k_bf, kmean, kmean_b);
    k_means<<<BS * NW, 256, 0, stream>>>(v_bf, nullptr, vmean_b);
    // 6. coarse similarity + top-8
    k_topk<<<BS * NW, 256, 0, stream>>>(qmean, kmean, top_idx);
    // 7. windowed attention (overwrites s_bf as msg_bf)
    k_attn<<<BS * NW * NH, 128, 0, stream>>>(q_bf, k_bf, v_bf, kmean_b, vmean_b,
                                             top_idx, msg_bf);
    // 8. msg @ Wm (window-major in -> pixel-major fp32 out)
    k_gemm<<<dim3(392, 4), 128, 0, stream>>>(msg_bf, D, Wm_bf, D, msgWm, D, D, w2p, 0);
    // 9. LN -> concat right half
    k_ln_concat<<<PIX, 256, 0, stream>>>(msgWm, g1, b1, cat);
    // 10. relu(concat @ W1) -> bf16
    k_gemm<<<dim3(392, 8), 128, 0, stream>>>(cat, 512, W1_bf, 512, hmid, 512, 512,
                                             nullptr, 1 | 2);
    // 11. hmid @ W2 -> fp32
    k_gemm<<<dim3(392, 4), 128, 0, stream>>>(hmid, 512, W2_bf, D, msg2, D, 512,
                                             nullptr, 0);
    // 12. LN + residual (reuse msgWm buffer for pre-transpose output)
    k_ln_res<<<PIX, 256, 0, stream>>>(msg2, xf, g2, b2, msgWm);
    // 13. back to NCHW
    k_transpose_out<<<tg, tb, 0, stream>>>(msgWm, out);
}
